// SparseAttention_14388140441621
// MI455X (gfx1250) — matrix operand order repaired and checked by reading, not compiled, not run on hardware
//
#include <hip/hip_runtime.h>

typedef __attribute__((ext_vector_type(16))) _Float16 v16h;
typedef __attribute__((ext_vector_type(8)))  float    v8f;

#define S_LEN   4096
#define D_DIM   64
#define H_NUM   16
#define HALF_W  128

// ---- DPP16 cross-lane helpers (stay within a 16-lane row; no LDS traffic) ----
template <int CTRL>
__device__ __forceinline__ float dpp_mov(float x) {
    return __builtin_bit_cast(float,
        __builtin_amdgcn_update_dpp(0, __builtin_bit_cast(int, x),
                                    CTRL, 0xf, 0xf, true));
}

__device__ __forceinline__ float row_max16(float x) {
    x = fmaxf(x, dpp_mov<0xB1>(x));   // quad_perm xor1
    x = fmaxf(x, dpp_mov<0x4E>(x));   // quad_perm xor2
    x = fmaxf(x, dpp_mov<0x141>(x));  // row_half_mirror
    x = fmaxf(x, dpp_mov<0x140>(x));  // row_mirror
    return x;
}

__device__ __forceinline__ float row_sum16(float x) {
    x += dpp_mov<0xB1>(x);
    x += dpp_mov<0x4E>(x);
    x += dpp_mov<0x141>(x);
    x += dpp_mov<0x140>(x);
    return x;
}

// ---- shared: mask + online softmax + C->A transpose via LDS; returns P frag --
template <bool MASKED, bool BOUND>
__device__ __forceinline__ v16h softmax_pa(
    const v8f (&sc)[2], int kb, int q0, int n, int hi,
    float (&mrun)[8], float (&lrun)[8], v8f (&acc)[4], _Float16* ldsP)
{
    __syncthreads();                     // WAR vs previous step's pa loads
    const int j0 = kb + n;
    const int j1 = kb + 16 + n;
    #pragma unroll
    for (int r = 0; r < 8; ++r) {
        float s0 = sc[0][r], s1 = sc[1][r];
        if (MASKED) {
            const int row = q0 + r + 8 * hi;   // C layout: VGPR r, half hi
            bool ok0 = ((unsigned)(j0 - row + HALF_W) < 2u * HALF_W);
            bool ok1 = ((unsigned)(j1 - row + HALF_W) < 2u * HALF_W);
            if (BOUND) {
                ok0 = ok0 && ((unsigned)j0 < (unsigned)S_LEN);
                ok1 = ok1 && ((unsigned)j1 < (unsigned)S_LEN);
            }
            s0 = ok0 ? s0 : -3.0e30f;
            s1 = ok1 ? s1 : -3.0e30f;
        }

        const float vmax  = row_max16(fmaxf(s0, s1));
        const float mnew  = fmaxf(mrun[r], vmax);
        const float alpha = __builtin_amdgcn_exp2f(mrun[r] - mnew);
        const float e0    = __builtin_amdgcn_exp2f(s0 - mnew);
        const float e1    = __builtin_amdgcn_exp2f(s1 - mnew);
        const float psum  = row_sum16(e0 + e1);

        lrun[r] = lrun[r] * alpha + psum;
        mrun[r] = mnew;
        #pragma unroll
        for (int fb = 0; fb < 4; ++fb) acc[fb][r] *= alpha;

        ldsP[(r + 8 * hi) * 32 + n]      = (_Float16)e0;
        ldsP[(r + 8 * hi) * 32 + 16 + n] = (_Float16)e1;
    }
    __syncthreads();

    v16h pa;
    const _Float16* base = &ldsP[n * 32];
    #pragma unroll
    for (int e = 0; e < 8; ++e) {
        pa[e]     = base[16 * hi + e];
        pa[e + 8] = base[16 * hi + 8 + e];
    }
    return pa;
}

// ============================ FAST PATH (f16 ws) =============================
// kf: K as f16, [H][S][D] row-major.  vt: V^T as f16, [H][D][S] row-major.
template <bool MASKED, bool BOUND>
__device__ __forceinline__ void attn_step_h(
    int kb, int q0, int n, int hi,
    const _Float16* __restrict__ kf, const _Float16* __restrict__ vt,
    const v16h (&qa)[2], float (&mrun)[8], float (&lrun)[8], v8f (&acc)[4],
    _Float16* ldsP)
{
    // ---- scores: each K B-fragment is one aligned 32B v16h load ----
    v8f sc[2];
    #pragma unroll
    for (int t = 0; t < 2; ++t) {
        const int key = kb + 16 * t + n;
        v8f c8 = (v8f){};
        #pragma unroll
        for (int c = 0; c < 2; ++c) {
            v16h b;
            if (!BOUND || (unsigned)key < (unsigned)S_LEN)
                b = *reinterpret_cast<const v16h*>(
                        kf + (size_t)key * D_DIM + 32 * c + 16 * hi);
            else
                b = (v16h){};
            c8 = __builtin_amdgcn_wmma_f32_16x16x32_f16(
                    false, qa[c], false, b, (short)0, c8, false, false);
        }
        sc[t] = c8;
    }

    const v16h pa = softmax_pa<MASKED, BOUND>(sc, kb, q0, n, hi,
                                              mrun, lrun, acc, ldsP);

    // ---- O += P V: V^T makes each B-fragment one aligned 32B v16h load ----
    const int kbase = kb + 16 * hi;      // 16-key chunk is whole-in or whole-out
    #pragma unroll
    for (int fb = 0; fb < 4; ++fb) {
        v16h vb;
        if (!BOUND || (unsigned)kbase < (unsigned)S_LEN)
            vb = *reinterpret_cast<const v16h*>(
                    vt + (size_t)(fb * 16 + n) * S_LEN + kbase);
        else
            vb = (v16h){};
        acc[fb] = __builtin_amdgcn_wmma_f32_16x16x32_f16(
                      false, pa, false, vb, (short)0, acc[fb], false, false);
    }
}

__global__ __launch_bounds__(32) void attn_fast(
    const float* __restrict__ q,
    const _Float16* __restrict__ kf,
    const _Float16* __restrict__ vt,
    float* __restrict__ out)
{
    __shared__ _Float16 ldsP[16 * 32];

    const int lane = threadIdx.x;
    const int n    = lane & 15;
    const int hi   = lane >> 4;

    const int tile = blockIdx.x;
    const int h    = tile >> 8;
    const int q0   = (tile & 255) << 4;

    const float*    qh  = q  + (size_t)h * S_LEN * D_DIM;
    const _Float16* kfh = kf + (size_t)h * S_LEN * D_DIM;
    const _Float16* vth = vt + (size_t)h * D_DIM * S_LEN;
    float*          oh  = out + (size_t)h * S_LEN * D_DIM;

    const float qscale = 0.125f * 1.44269504088896340736f; // 1/sqrt(64)*log2(e)

    v16h qa[2];
    {
        const float* qrow = qh + (size_t)(q0 + n) * D_DIM;
        #pragma unroll
        for (int c = 0; c < 2; ++c) {
            const float* s0p = qrow + 32 * c + 16 * hi;
            const float* s1p = qrow + 32 * c + 16 * hi + 8;
            #pragma unroll
            for (int e = 0; e < 8; ++e) {
                qa[c][e]     = (_Float16)(s0p[e] * qscale);
                qa[c][e + 8] = (_Float16)(s1p[e] * qscale);
            }
        }
    }

    float mrun[8], lrun[8];
    v8f   acc[4];
    #pragma unroll
    for (int r = 0; r < 8; ++r) { mrun[r] = -1e30f; lrun[r] = 0.0f; }
    #pragma unroll
    for (int fb = 0; fb < 4; ++fb) acc[fb] = (v8f){};

    const int jstart = q0 - HALF_W;

    if (q0 >= HALF_W && q0 + 160 <= S_LEN) {
        attn_step_h<true, false>(jstart, q0, n, hi, kfh, vth, qa, mrun, lrun, acc, ldsP);
        #pragma unroll 1
        for (int s = 1; s <= 7; ++s) {
            const int kb = jstart + 32 * s;
            {   // prefetch next step (global_prefetch_b8)
                const int pk = kb + 32 + n + 16 * hi;
                if (pk < S_LEN)
                    __builtin_prefetch(kfh + (size_t)pk * D_DIM, 0, 1);
                if (kb + 64 < S_LEN) {
                    __builtin_prefetch(vth + (size_t)(n + 16 * hi) * S_LEN + kb + 64, 0, 1);
                    __builtin_prefetch(vth + (size_t)(n + 16 * hi + 32) * S_LEN + kb + 64, 0, 1);
                }
            }
            attn_step_h<false, false>(kb, q0, n, hi, kfh, vth, qa, mrun, lrun, acc, ldsP);
        }
        attn_step_h<true, false>(jstart + 256, q0, n, hi, kfh, vth, qa, mrun, lrun, acc, ldsP);
    } else {
        #pragma unroll 1
        for (int kb = jstart; kb < q0 + 16 + HALF_W; kb += 32)
            attn_step_h<true, true>(kb, q0, n, hi, kfh, vth, qa, mrun, lrun, acc, ldsP);
    }

    #pragma unroll
    for (int r = 0; r < 8; ++r) {
        const float inv = 1.0f / lrun[r];
        float* orow = oh + (size_t)(q0 + r + 8 * hi) * D_DIM;
        #pragma unroll
        for (int fb = 0; fb < 4; ++fb)
            orow[fb * 16 + n] = acc[fb][r] * inv;
    }
}

// ---- one-shot pass: K -> f16 (same layout), V -> f16 transposed [H][D][S] ---
__global__ __launch_bounds__(256) void convert_kv(
    const float* __restrict__ k, const float* __restrict__ v,
    _Float16* __restrict__ kf, _Float16* __restrict__ vt)
{
    __shared__ _Float16 tile[64 * 65];   // padded 64x64 transpose tile

    const int blk = blockIdx.x;          // H_NUM * (S/64) blocks
    const int h   = blk >> 6;
    const int k0  = (blk & 63) << 6;

    const float* kh = k + ((size_t)h * S_LEN + k0) * D_DIM;
    const float* vh = v + ((size_t)h * S_LEN + k0) * D_DIM;
    _Float16* kfh   = kf + ((size_t)h * S_LEN + k0) * D_DIM;
    _Float16* vth   = vt + (size_t)h * D_DIM * S_LEN;

    const int t  = threadIdx.x;
    const int f  = t & 63;
    const int r0 = t >> 6;               // 0..3
    #pragma unroll 1
    for (int rr = r0; rr < 64; rr += 4) {
        const float kvv = kh[(size_t)rr * D_DIM + f];   // coalesced
        const float vvv = vh[(size_t)rr * D_DIM + f];   // coalesced
        kfh[(size_t)rr * D_DIM + f] = (_Float16)kvv;    // coalesced
        tile[f * 65 + rr] = (_Float16)vvv;              // transpose in LDS
    }
    __syncthreads();

    const int fo = t >> 2;               // 0..63 : feature row of V^T
    const int c  = t & 3;                // 16-key chunk
    _Float16* dst = vth + (size_t)fo * S_LEN + k0 + c * 16;
    const _Float16* src = &tile[fo * 65 + c * 16];
    #pragma unroll
    for (int e = 0; e < 16; ++e) dst[e] = src[e];       // coalesced across wave
}

// ======================= FALLBACK PATH (fp32 direct) =========================
template <bool MASKED, bool BOUND>
__device__ __forceinline__ void attn_step_f32(
    int kb, int q0, int n, int hi,
    const float* __restrict__ kh, const float* __restrict__ vh,
    const v16h (&qa)[2], float (&mrun)[8], float (&lrun)[8], v8f (&acc)[4],
    _Float16* ldsP)
{
    v8f sc[2];
    #pragma unroll
    for (int t = 0; t < 2; ++t) {
        const int key = kb + 16 * t + n;
        v8f c8 = (v8f){};
        #pragma unroll
        for (int c = 0; c < 2; ++c) {
            v16h b;
            if (!BOUND || (unsigned)key < (unsigned)S_LEN) {
                const float* kp = kh + (size_t)key * D_DIM + 32 * c + 16 * hi;
                #pragma unroll
                for (int e = 0; e < 16; ++e) b[e] = (_Float16)kp[e];
            } else {
                #pragma unroll
                for (int e = 0; e < 16; ++e) b[e] = (_Float16)0.0f;
            }
            c8 = __builtin_amdgcn_wmma_f32_16x16x32_f16(
                    false, qa[c], false, b, (short)0, c8, false, false);
        }
        sc[t] = c8;
    }

    const v16h pa = softmax_pa<MASKED, BOUND>(sc, kb, q0, n, hi,
                                              mrun, lrun, acc, ldsP);

    #pragma unroll
    for (int fb = 0; fb < 4; ++fb) {
        v16h vb;
        #pragma unroll
        for (int e = 0; e < 16; ++e) {
            const int key = kb + 16 * hi + e;
            float val = 0.0f;
            if (!BOUND || (unsigned)key < (unsigned)S_LEN)
                val = vh[(size_t)key * D_DIM + fb * 16 + n];
            vb[e] = (_Float16)val;
        }
        acc[fb] = __builtin_amdgcn_wmma_f32_16x16x32_f16(
                      false, pa, false, vb, (short)0, acc[fb], false, false);
    }
}

__global__ __launch_bounds__(32) void attn_f32(
    const float* __restrict__ q,
    const float* __restrict__ k,
    const float* __restrict__ v,
    float* __restrict__ out)
{
    __shared__ _Float16 ldsP[16 * 32];

    const int lane = threadIdx.x;
    const int n    = lane & 15;
    const int hi   = lane >> 4;

    const int tile = blockIdx.x;
    const int h    = tile >> 8;
    const int q0   = (tile & 255) << 4;

    const float* qh = q + (size_t)h * S_LEN * D_DIM;
    const float* kh = k + (size_t)h * S_LEN * D_DIM;
    const float* vh = v + (size_t)h * S_LEN * D_DIM;
    float*       oh = out + (size_t)h * S_LEN * D_DIM;

    const float qscale = 0.125f * 1.44269504088896340736f;

    v16h qa[2];
    {
        const float* qrow = qh + (size_t)(q0 + n) * D_DIM;
        #pragma unroll
        for (int c = 0; c < 2; ++c) {
            const float* s0p = qrow + 32 * c + 16 * hi;
            const float* s1p = qrow + 32 * c + 16 * hi + 8;
            #pragma unroll
            for (int e = 0; e < 8; ++e) {
                qa[c][e]     = (_Float16)(s0p[e] * qscale);
                qa[c][e + 8] = (_Float16)(s1p[e] * qscale);
            }
        }
    }

    float mrun[8], lrun[8];
    v8f   acc[4];
    #pragma unroll
    for (int r = 0; r < 8; ++r) { mrun[r] = -1e30f; lrun[r] = 0.0f; }
    #pragma unroll
    for (int fb = 0; fb < 4; ++fb) acc[fb] = (v8f){};

    const int jstart = q0 - HALF_W;

    if (q0 >= HALF_W && q0 + 160 <= S_LEN) {
        attn_step_f32<true, false>(jstart, q0, n, hi, kh, vh, qa, mrun, lrun, acc, ldsP);
        #pragma unroll 1
        for (int s = 1; s <= 7; ++s) {
            const int kb = jstart + 32 * s;
            attn_step_f32<false, false>(kb, q0, n, hi, kh, vh, qa, mrun, lrun, acc, ldsP);
        }
        attn_step_f32<true, false>(jstart + 256, q0, n, hi, kh, vh, qa, mrun, lrun, acc, ldsP);
    } else {
        #pragma unroll 1
        for (int kb = jstart; kb < q0 + 16 + HALF_W; kb += 32)
            attn_step_f32<true, true>(kb, q0, n, hi, kh, vh, qa, mrun, lrun, acc, ldsP);
    }

    #pragma unroll
    for (int r = 0; r < 8; ++r) {
        const float inv = 1.0f / lrun[r];
        float* orow = oh + (size_t)(q0 + r + 8 * hi) * D_DIM;
        #pragma unroll
        for (int fb = 0; fb < 4; ++fb)
            orow[fb * 16 + n] = acc[fb][r] * inv;
    }
}

extern "C" void kernel_launch(void* const* d_in, const int* in_sizes, int n_in,
                              void* d_out, int out_size, void* d_ws, size_t ws_size,
                              hipStream_t stream) {
    (void)in_sizes; (void)n_in; (void)out_size;
    const float* q = (const float*)d_in[0];
    const float* k = (const float*)d_in[1];
    const float* v = (const float*)d_in[2];
    float* out = (float*)d_out;

    const size_t elems  = (size_t)H_NUM * S_LEN * D_DIM;      // 4,194,304
    const size_t fbytes = elems * sizeof(_Float16);           // 8 MB each

    if (ws_size >= 2 * fbytes) {
        _Float16* kf = (_Float16*)d_ws;
        _Float16* vt = kf + elems;
        convert_kv<<<H_NUM * (S_LEN / 64), 256, 0, stream>>>(k, v, kf, vt);
        attn_fast<<<H_NUM * (S_LEN / 16), 32, 0, stream>>>(q, kf, vt, out);
    } else {
        attn_f32<<<H_NUM * (S_LEN / 16), 32, 0, stream>>>(q, k, v, out);
    }
}
